// NRI_15582141349981
// MI455X (gfx1250) — matrix operand order repaired and checked by reading, not compiled, not run on hardware
//
#include <hip/hip_runtime.h>
#include <hip/hip_bf16.h>
#include <math.h>

// ---------------- problem constants ----------------
#define NN 10000
#define NE 160000
// NF=48, HID=NED=128, K=4, SEQ=8, PRED=2, TAU=0.5, EPS=1e-5

typedef __attribute__((ext_vector_type(16))) _Float16 v16h;
typedef __attribute__((ext_vector_type(8)))  _Float16 v8h;
typedef __attribute__((ext_vector_type(8)))  float    v8f;

__device__ __forceinline__ v8f wmma32(v16h a, v16h b, v8f c) {
  // D = A(16x32 f16) * B(32x16 f16) + C(16x16 f32)
  return __builtin_amdgcn_wmma_f32_16x16x32_f16(false, a, false, b, (short)0, c, false, false);
}

__device__ __forceinline__ v8f splat8(float v) {
  v8f r = {v, v, v, v, v, v, v, v};
  return r;
}

// Wave-local LDS ordering: DS ops are in-order per wave; we only need to stop
// the compiler from reordering and to wait DScnt before consuming.
__device__ __forceinline__ void lds_fence() {
  __builtin_amdgcn_fence(__ATOMIC_RELEASE, "workgroup");
  __builtin_amdgcn_wave_barrier();
  __builtin_amdgcn_fence(__ATOMIC_ACQUIRE, "workgroup");
}

__device__ __forceinline__ v16h cat8(v8h lo, v8h hi) {
  return __builtin_shufflevector(lo, hi, 0, 1, 2, 3, 4, 5, 6, 7, 8, 9, 10, 11, 12, 13, 14, 15);
}

// ---- A fragment (16x32 f16): lane m=lane&15 is row M, hf=lane>>4 picks K-halves.
// Per lane the 16 values are exactly two contiguous 8xf16 (16B) runs:
//   K in [8*hf, 8*hf+7] and [16+8*hf, 23+8*hf]  ->  2x b128 loads, no packing.
__device__ __forceinline__ v16h a_frag(const _Float16* row, int hf) {
  v8h lo = *(const v8h*)(row + (hf << 3));
  v8h hi = *(const v8h*)(row + 16 + (hf << 3));
  return cat8(lo, hi);
}

// Zero-padded run load (runs never straddle kmax here: kmax % 8 == 0).
// With compile-time k/kmax this folds to either a plain load or a zero vector.
__device__ __forceinline__ v8h ld8z(const _Float16* r, int k, int kmax) {
  if (k + 8 <= kmax) return *(const v8h*)(r + k);
  v8h z = {};
  return z;
}

__device__ __forceinline__ v16h a_frag_pad(const _Float16* row, int colbase, int kmax, int hf) {
  v8h lo = ld8z(row, colbase + (hf << 3), kmax);
  v8h hi = ld8z(row, colbase + 16 + (hf << 3), kmax);
  return cat8(lo, hi);
}

// Run load over concat([rowA(48), rowB(48)]): 48 % 8 == 0 so a run never straddles.
__device__ __forceinline__ v8h ld8cat(const _Float16* ra, const _Float16* rb, int k) {
  const _Float16* p = (k < 48) ? (ra + k) : (rb + (k - 48));
  return *(const v8h*)p;
}

__device__ __forceinline__ v16h a_frag_cat48(const _Float16* ra, const _Float16* rb,
                                             int colbase, int hf) {
  v8h lo = ld8cat(ra, rb, colbase + (hf << 3));
  v8h hi = ld8cat(ra, rb, colbase + 16 + (hf << 3));
  return cat8(lo, hi);
}

// ---- B fragment (32x16 f16) from PRE-SWIZZLED weights.
// Swizzled layout: for tile (c,ct), lane l = hf*16+n stores its 16 table-order K
// values contiguously: swz[(tid*32+l)*16 + j] = W[(c*32 + hf*8 + j + (j >= 8 ? 8 : 0))*ld + ct*16 + n].
// Fragment load = one 32B contiguous region per lane (2x b128).
__device__ __forceinline__ v16h b_frag_sw(const _Float16* Wsw, int tid, int lane) {
  const _Float16* p = Wsw + (((size_t)tid * 32 + lane) << 4);
  v8h lo = *(const v8h*)p;
  v8h hi = *(const v8h*)(p + 8);
  return cat8(lo, hi);
}

// One K-chunk of a 16xN tile GEMM: preload all NCT B-fragments (load clause),
// then issue NCT WMMAs -- lets the scheduler hide load latency behind WMMAs
// instead of serializing on s_wait_loadcnt 0.
template <int NCT>
__device__ __forceinline__ void gemm_chunk(v8f* acc, v16h af, const _Float16* Wsw,
                                           int tidbase, int lane) {
  v16h bf[NCT];
#pragma unroll
  for (int ct = 0; ct < NCT; ++ct) bf[ct] = b_frag_sw(Wsw, tidbase + ct, lane);
#pragma unroll
  for (int ct = 0; ct < NCT; ++ct) acc[ct] = wmma32(af, bf[ct], acc[ct]);
}

// ================= weight swizzle (f32 -> f16 fragment-major) =================
// One thread per (tile, lane): writes 16 f16. Zero-fills rows k >= Kreal (padding).
__global__ void k_swizzle_b(const float* __restrict__ W, _Float16* __restrict__ out,
                            int Kreal, int Ktot, int Ncols) {
  int idx = blockIdx.x * blockDim.x + threadIdx.x;
  int nct = Ncols >> 4;
  int total = (Ktot >> 5) * nct * 32;
  if (idx >= total) return;
  int lane = idx & 31, tid = idx >> 5;
  int c = tid / nct, ct = tid - c * nct;
  int hf = lane >> 4, n = lane & 15;
  _Float16* o = out + (size_t)idx * 16;
#pragma unroll
  for (int j = 0; j < 16; ++j) {
    int k = c * 32 + (j < 8 ? hf * 8 + j : 16 + hf * 8 + (j - 8));
    float v = (k < Kreal) ? W[(size_t)k * Ncols + ct * 16 + n] : 0.f;
    o[j] = (_Float16)v;
  }
}

// ================= generic elementwise kernels =================
__global__ void k_f32_to_f16(const float* __restrict__ s, _Float16* __restrict__ d, int n) {
  int i = blockIdx.x * blockDim.x + threadIdx.x;
  if (i < n) d[i] = (_Float16)s[i];
}

__global__ void k_bn_finalize(const float* __restrict__ ssum, const float* __restrict__ ssq,
                              const float* __restrict__ g, const float* __restrict__ be,
                              float count, float* __restrict__ scale, float* __restrict__ shift) {
  int c = threadIdx.x; // 128
  float mean = ssum[c] / count;
  float var  = ssq[c] / count - mean * mean;
  float sc   = g[c] * rsqrtf(var + 1e-5f);
  scale[c] = sc;
  shift[c] = be[c] - mean * sc;
}

__global__ void k_bn_apply(_Float16* __restrict__ y, const float* __restrict__ scale,
                           const float* __restrict__ shift, int total) {
  int i = blockIdx.x * blockDim.x + threadIdx.x;
  if (i < total) {
    int c = i & 127;
    y[i] = (_Float16)(scale[c] * (float)y[i] + shift[c]);
  }
}

// ================= WMMA MLP over row tiles (encoder K1=48 / m7 K1=128) =================
// y = relu(relu(A@W1+b1)@W2+b2), y stored f16 (R x 128); per-channel sum/sumsq atomics.
// K1 is compile-time: pad checks and chunk loop fold statically (no exec-mask branches).
template <int K1>
__global__ void __launch_bounds__(256)
k_mlp2(const _Float16* __restrict__ A, int lda,
       const _Float16* __restrict__ W1, const float* __restrict__ b1,
       const _Float16* __restrict__ W2, const float* __restrict__ b2,
       _Float16* __restrict__ out, float* __restrict__ ssum, float* __restrict__ ssq, int R) {
  __shared__ alignas(16) _Float16 stg[8][16 * 128];
  int wave = threadIdx.x >> 5, lane = threadIdx.x & 31;
  int hf = lane >> 4, nc = lane & 15, mr = lane & 15;
  int tile = blockIdx.x * 8 + wave;
  if (tile * 16 >= R) return;
  _Float16* S = &stg[wave][0];
  const _Float16* arow = A + (size_t)(tile * 16 + mr) * lda;

  v8f acc[8];
#pragma unroll
  for (int ct = 0; ct < 8; ++ct) acc[ct] = splat8(b1[ct * 16 + nc]);
  constexpr int NCH = (K1 + 31) / 32;
#pragma unroll
  for (int c = 0; c < NCH; ++c) {
    v16h af = a_frag_pad(arow, c * 32, K1, hf);
    gemm_chunk<8>(acc, af, W1, c * 8, lane);
  }
#pragma unroll
  for (int ct = 0; ct < 8; ++ct)
#pragma unroll
    for (int i = 0; i < 8; ++i) {
      float v = acc[ct][i]; v = v > 0.f ? v : 0.f;
      S[(hf * 8 + i) * 128 + ct * 16 + nc] = (_Float16)v;
    }
  lds_fence();
#pragma unroll
  for (int ct = 0; ct < 8; ++ct) acc[ct] = splat8(b2[ct * 16 + nc]);
#pragma unroll
  for (int c = 0; c < 4; ++c) {
    v16h af = a_frag(S + mr * 128 + c * 32, hf);
    gemm_chunk<8>(acc, af, W2, c * 8, lane);
  }
#pragma unroll
  for (int ct = 0; ct < 8; ++ct) {
    float s = 0.f, s2 = 0.f;
#pragma unroll
    for (int i = 0; i < 8; ++i) {
      float v = acc[ct][i]; v = v > 0.f ? v : 0.f;
      out[(size_t)(tile * 16 + hf * 8 + i) * 128 + ct * 16 + nc] = (_Float16)v;
      s += v; s2 += v * v;
    }
    atomicAdd(&ssum[ct * 16 + nc], s);
    atomicAdd(&ssq[ct * 16 + nc], s2);
  }
}

// ================= WMMA edge MLP (m6: NSEG=2 K=256, m8: NSEG=4 K=512) =================
template <int NSEG>
__global__ void __launch_bounds__(256)
k_edge_mlp(const _Float16* __restrict__ s0, const int* __restrict__ i0,
           const _Float16* __restrict__ s1, const int* __restrict__ i1,
           const _Float16* __restrict__ s2, const int* __restrict__ i2,
           const _Float16* __restrict__ s3, const int* __restrict__ i3,
           const _Float16* __restrict__ W1, const float* __restrict__ b1,
           const _Float16* __restrict__ W2, const float* __restrict__ b2,
           _Float16* __restrict__ out, float* __restrict__ ssum, float* __restrict__ ssq, int Etot) {
  __shared__ alignas(16) _Float16 stg[8][16 * 128];
  int wave = threadIdx.x >> 5, lane = threadIdx.x & 31;
  int hf = lane >> 4, nc = lane & 15, mr = lane & 15;
  int tile = blockIdx.x * 8 + wave;
  if (tile * 16 >= Etot) return;
  _Float16* S = &stg[wave][0];
  const _Float16* segs[4] = {s0, s1, s2, s3};
  const int* idxs[4] = {i0, i1, i2, i3};

  v8f acc[8];
#pragma unroll
  for (int ct = 0; ct < 8; ++ct) acc[ct] = splat8(b1[ct * 16 + nc]);
#pragma unroll
  for (int sg = 0; sg < NSEG; ++sg) {
    const _Float16* base = segs[sg] + (size_t)idxs[sg][tile * 16 + mr] * 128;
#pragma unroll
    for (int cc = 0; cc < 4; ++cc) {
      v16h af = a_frag(base + cc * 32, hf);
      gemm_chunk<8>(acc, af, W1, (sg * 4 + cc) * 8, lane);
    }
  }
#pragma unroll
  for (int ct = 0; ct < 8; ++ct)
#pragma unroll
    for (int i = 0; i < 8; ++i) {
      float v = acc[ct][i]; v = v > 0.f ? v : 0.f;
      S[(hf * 8 + i) * 128 + ct * 16 + nc] = (_Float16)v;
    }
  lds_fence();
#pragma unroll
  for (int ct = 0; ct < 8; ++ct) acc[ct] = splat8(b2[ct * 16 + nc]);
#pragma unroll
  for (int c = 0; c < 4; ++c) {
    v16h af = a_frag(S + mr * 128 + c * 32, hf);
    gemm_chunk<8>(acc, af, W2, c * 8, lane);
  }
#pragma unroll
  for (int ct = 0; ct < 8; ++ct) {
    float s = 0.f, s2 = 0.f;
#pragma unroll
    for (int i = 0; i < 8; ++i) {
      float v = acc[ct][i]; v = v > 0.f ? v : 0.f;
      out[(size_t)(tile * 16 + hf * 8 + i) * 128 + ct * 16 + nc] = (_Float16)v;
      s += v; s2 += v * v;
    }
    atomicAdd(&ssum[ct * 16 + nc], s);
    atomicAdd(&ssq[ct * 16 + nc], s2);
  }
}

// ================= BN(msg) + segment scatter into agg =================
__global__ void k_msg_scatter(const _Float16* __restrict__ msgpre, const float* __restrict__ scale,
                              const float* __restrict__ shift, const int* __restrict__ tgt,
                              float* __restrict__ agg, int Etot) {
  size_t i = (size_t)blockIdx.x * blockDim.x + threadIdx.x;
  if (i < (size_t)Etot * 128) {
    int c = (int)(i & 127);
    size_t e = i >> 7;
    float v = scale[c] * (float)msgpre[i] + shift[c];
    atomicAdd(&agg[(size_t)tgt[e] * 128 + c], v);
  }
}

// ================= fold BN into logits weights, logits/softmax/zk =================
__global__ void k_fold_w3(const float* __restrict__ w3, const float* __restrict__ b3,
                          const float* __restrict__ scale, const float* __restrict__ shift,
                          float* __restrict__ w3f, float* __restrict__ b3f) {
  int c = threadIdx.x; // 128
#pragma unroll
  for (int j = 0; j < 4; ++j) w3f[c * 4 + j] = scale[c] * w3[c * 4 + j];
  if (c < 4) {
    float acc = b3[c];
    for (int r = 0; r < 128; ++r) acc += shift[r] * w3[r * 4 + c];
    b3f[c] = acc;
  }
}

__global__ void k_logits_zk(const _Float16* __restrict__ h8, const float* __restrict__ w3f,
                            const float* __restrict__ b3f, const float* __restrict__ gum,
                            float* __restrict__ zk, int Etot) {
  int e = blockIdx.x * blockDim.x + threadIdx.x;
  if (e >= Etot) return;
  float l0 = b3f[0], l1 = b3f[1], l2 = b3f[2], l3 = b3f[3];
  const _Float16* hp = h8 + (size_t)e * 128;
  for (int c = 0; c < 128; ++c) {
    float hv = (float)hp[c];
    l0 += hv * w3f[c * 4 + 0];
    l1 += hv * w3f[c * 4 + 1];
    l2 += hv * w3f[c * 4 + 2];
    l3 += hv * w3f[c * 4 + 3];
  }
  l0 = (l0 + gum[(size_t)e * 4 + 0]) * 2.f; // /TAU, TAU=0.5
  l1 = (l1 + gum[(size_t)e * 4 + 1]) * 2.f;
  l2 = (l2 + gum[(size_t)e * 4 + 2]) * 2.f;
  l3 = (l3 + gum[(size_t)e * 4 + 3]) * 2.f;
  float mx = fmaxf(fmaxf(l0, l1), fmaxf(l2, l3));
  float e0 = expf(l0 - mx), e1 = expf(l1 - mx), e2 = expf(l2 - mx), e3 = expf(l3 - mx);
  float inv = 1.f / (e0 + e1 + e2 + e3);
  zk[(size_t)e * 3 + 0] = e1 * inv;
  zk[(size_t)e * 3 + 1] = e2 * inv;
  zk[(size_t)e * 3 + 2] = e3 * inv;
}

__global__ void k_deg(const int* __restrict__ tgt, float* __restrict__ deg, int Etot) {
  int e = blockIdx.x * blockDim.x + threadIdx.x;
  if (e < Etot) atomicAdd(&deg[tgt[e]], 1.f);
}

// ================= recurrent-step kernels =================
__global__ void k_prep_inp(const float* __restrict__ x, const float* __restrict__ outbuf,
                           float* __restrict__ inp, int t, int Nn) {
  int i = blockIdx.x * blockDim.x + threadIdx.x;
  if (i < Nn * 6) {
    int n = i / 6, j = i % 6;
    inp[i] = (t < 6) ? x[(size_t)n * 48 + t * 6 + j] : outbuf[(size_t)n * 48 + (t - 1) * 6 + j];
  }
}

// xo = relu(h @ rec_w(128x48) + rec_b)  (WMMA, 3 col-tiles, Ncols=48 -> nct=3)
__global__ void __launch_bounds__(256)
k_xo(const _Float16* __restrict__ h, const _Float16* __restrict__ Wr,
     const float* __restrict__ br, _Float16* __restrict__ xo, int Nn) {
  int wave = threadIdx.x >> 5, lane = threadIdx.x & 31;
  int hf = lane >> 4, nc = lane & 15, mr = lane & 15;
  int tile = blockIdx.x * 8 + wave;
  if (tile * 16 >= Nn) return;
  v8f acc[3];
#pragma unroll
  for (int ct = 0; ct < 3; ++ct) acc[ct] = splat8(br[ct * 16 + nc]);
  const _Float16* arow = h + (size_t)(tile * 16 + mr) * 128;
#pragma unroll
  for (int c = 0; c < 4; ++c) {
    v16h af = a_frag(arow + c * 32, hf);
    gemm_chunk<3>(acc, af, Wr, c * 3, lane);
  }
#pragma unroll
  for (int ct = 0; ct < 3; ++ct)
#pragma unroll
    for (int i = 0; i < 8; ++i) {
      float v = acc[ct][i]; v = v > 0.f ? v : 0.f;
      xo[(size_t)(tile * 16 + hf * 8 + i) * 48 + ct * 16 + nc] = (_Float16)v;
    }
}

// Per-edge 3x MLP(96->128->128), zk-weighted sum, atomic scatter to magg[tgt]
__global__ void __launch_bounds__(256)
k_edge_recur(const _Float16* __restrict__ xo, const int* __restrict__ src,
             const int* __restrict__ tgt, const _Float16* __restrict__ We1,
             const float* __restrict__ be1, const _Float16* __restrict__ We2,
             const float* __restrict__ be2, const float* __restrict__ zk,
             float* __restrict__ magg, int Etot) {
  __shared__ alignas(16) _Float16 stg[8][16 * 128];
  __shared__ float szk[8][16 * 3];
  __shared__ int stgt[8][16];
  int wave = threadIdx.x >> 5, lane = threadIdx.x & 31;
  int hf = lane >> 4, nc = lane & 15, mr = lane & 15;
  int tile = blockIdx.x * 8 + wave;
  if (tile * 16 >= Etot) return;
  int e0 = tile * 16;
  if (lane < 16) {
    stgt[wave][lane] = tgt[e0 + lane];
#pragma unroll
    for (int k = 0; k < 3; ++k) szk[wave][lane * 3 + k] = zk[(size_t)(e0 + lane) * 3 + k];
  }
  const _Float16* rowT = xo + (size_t)tgt[e0 + mr] * 48;
  const _Float16* rowS = xo + (size_t)src[e0 + mr] * 48;
  lds_fence();
  _Float16* S = &stg[wave][0];
  v8f emsg[8];
#pragma unroll
  for (int ct = 0; ct < 8; ++ct) emsg[ct] = splat8(0.f);

  for (int k = 0; k < 3; ++k) {
    v8f acc[8];
#pragma unroll
    for (int ct = 0; ct < 8; ++ct) acc[ct] = splat8(be1[k * 128 + ct * 16 + nc]);
#pragma unroll
    for (int c = 0; c < 3; ++c) {
      v16h af = a_frag_cat48(rowT, rowS, c * 32, hf);
      gemm_chunk<8>(acc, af, We1 + (size_t)k * 96 * 128, c * 8, lane);
    }
#pragma unroll
    for (int ct = 0; ct < 8; ++ct)
#pragma unroll
      for (int i = 0; i < 8; ++i) {
        float v = acc[ct][i]; v = v > 0.f ? v : 0.f;
        S[(hf * 8 + i) * 128 + ct * 16 + nc] = (_Float16)v;
      }
    lds_fence();
#pragma unroll
    for (int ct = 0; ct < 8; ++ct) acc[ct] = splat8(be2[k * 128 + ct * 16 + nc]);
#pragma unroll
    for (int c = 0; c < 4; ++c) {
      v16h af = a_frag(S + mr * 128 + c * 32, hf);
      gemm_chunk<8>(acc, af, We2 + (size_t)k * 128 * 128, c * 8, lane);
    }
#pragma unroll
    for (int ct = 0; ct < 8; ++ct)
#pragma unroll
      for (int i = 0; i < 8; ++i)
        emsg[ct][i] += szk[wave][(hf * 8 + i) * 3 + k] * acc[ct][i];
    lds_fence(); // staging reused next k
  }
#pragma unroll
  for (int ct = 0; ct < 8; ++ct)
#pragma unroll
    for (int i = 0; i < 8; ++i)
      atomicAdd(&magg[(size_t)stgt[wave][hf * 8 + i] * 128 + ct * 16 + nc], emsg[ct][i]);
}

__global__ void k_mh(const float* __restrict__ magg, const float* __restrict__ deg,
                     _Float16* __restrict__ mh, int total) {
  int i = blockIdx.x * blockDim.x + threadIdx.x;
  if (i < total) {
    int n = i >> 7;
    float d = deg[n]; d = d > 1.f ? d : 1.f;
    mh[i] = (_Float16)(magg[i] / d);
  }
}

// GRU: gh = m @ gru_wh(128x384) via WMMA (24 col-tiles, in 3 groups of 8); gi inline (K=6)
__global__ void __launch_bounds__(256)
k_gru(const _Float16* __restrict__ mh, const _Float16* __restrict__ Wh,
      const float* __restrict__ bh, const float* __restrict__ wi, const float* __restrict__ bi,
      const float* __restrict__ inp, _Float16* __restrict__ hout, int Nn) {
  __shared__ float sinp[8][16 * 6];
  int wave = threadIdx.x >> 5, lane = threadIdx.x & 31;
  int hf = lane >> 4, nc = lane & 15, mr = lane & 15;
  int tile = blockIdx.x * 8 + wave;
  if (tile * 16 >= Nn) return;
  if (lane < 16) {
#pragma unroll
    for (int j = 0; j < 6; ++j) sinp[wave][lane * 6 + j] = inp[(size_t)(tile * 16 + lane) * 6 + j];
  }
  lds_fence();
  v8f acc[24];
#pragma unroll
  for (int ct = 0; ct < 24; ++ct) acc[ct] = splat8(bh[ct * 16 + nc]);
  const _Float16* arow = mh + (size_t)(tile * 16 + mr) * 128;
#pragma unroll
  for (int c = 0; c < 4; ++c) {
    v16h af = a_frag(arow + c * 32, hf);
#pragma unroll
    for (int g = 0; g < 3; ++g)
      gemm_chunk<8>(acc + g * 8, af, Wh, c * 24 + g * 8, lane);
  }
#pragma unroll
  for (int ct = 0; ct < 8; ++ct)
#pragma unroll
    for (int i = 0; i < 8; ++i) {
      int rl = hf * 8 + i;
      int grow = tile * 16 + rl;
      int col = ct * 16 + nc;
      float gr = bi[col], gz = bi[col + 128], gn = bi[col + 256];
#pragma unroll
      for (int j = 0; j < 6; ++j) {
        float xv = sinp[wave][rl * 6 + j];
        gr += xv * wi[j * 384 + col];
        gz += xv * wi[j * 384 + col + 128];
        gn += xv * wi[j * 384 + col + 256];
      }
      float r  = 1.f / (1.f + expf(-(gr + acc[ct][i])));
      float u  = 1.f / (1.f + expf(-(gz + acc[ct + 8][i])));
      float ng = tanhf(gn + r * acc[ct + 16][i]);
      float mv = (float)mh[(size_t)grow * 128 + col];
      float hv = (1.f - u) * ng + u * mv;
      hout[(size_t)grow * 128 + col] = (_Float16)hv;
    }
}

__global__ void k_mu(const _Float16* __restrict__ h, const float* __restrict__ dw,
                     const float* __restrict__ db, const float* __restrict__ inp,
                     float* __restrict__ outbuf, int t, int Nn) {
  int i = blockIdx.x * blockDim.x + threadIdx.x;
  if (i < Nn * 6) {
    int n = i / 6, j = i % 6;
    float a = db[j];
    const _Float16* hr = h + (size_t)n * 128;
    for (int c = 0; c < 128; ++c) a += (float)hr[c] * dw[c * 6 + j];
    a = a > 0.f ? a : 0.f;
    outbuf[(size_t)n * 48 + t * 6 + j] = inp[i] + a;
  }
}

// =================================================================
extern "C" void kernel_launch(void* const* d_in, const int* in_sizes, int n_in,
                              void* d_out, int out_size, void* d_ws, size_t ws_size,
                              hipStream_t stream) {
  (void)in_sizes; (void)n_in; (void)out_size; (void)ws_size;
  const int N = NN, E = NE;

  const float* x   = (const float*)d_in[0];
  const float* gum = (const float*)d_in[1];
  const float* P[38];
  for (int i = 0; i < 38; ++i) P[i] = (const float*)d_in[2 + i];
  const int* ei  = (const int*)d_in[40];
  const int* src = ei;
  const int* tgt = ei + E;
  float* out = (float*)d_out;

  // ---- workspace carve-out ----
  char* p = (char*)d_ws;
  auto alloc = [&](size_t bytes) -> char* {
    char* r = p;
    p += (bytes + 255) & ~(size_t)255;
    return r;
  };
  _Float16* wh_enc1 = (_Float16*)alloc(64 * 128 * 2); // K padded 48->64 (zero-filled)
  _Float16* wh_enc2 = (_Float16*)alloc(128 * 128 * 2);
  _Float16* wh_m61  = (_Float16*)alloc(256 * 128 * 2);
  _Float16* wh_m62  = (_Float16*)alloc(128 * 128 * 2);
  _Float16* wh_m71  = (_Float16*)alloc(128 * 128 * 2);
  _Float16* wh_m72  = (_Float16*)alloc(128 * 128 * 2);
  _Float16* wh_m81  = (_Float16*)alloc(512 * 128 * 2);
  _Float16* wh_m82  = (_Float16*)alloc(128 * 128 * 2);
  _Float16* wh_e1   = (_Float16*)alloc(3 * 96 * 128 * 2);
  _Float16* wh_e2   = (_Float16*)alloc(3 * 128 * 128 * 2);
  _Float16* wh_rec  = (_Float16*)alloc(128 * 48 * 2);
  _Float16* wh_gwh  = (_Float16*)alloc(128 * 384 * 2);
  _Float16* xh      = (_Float16*)alloc((size_t)N * 48 * 2);
  _Float16* h5h     = (_Float16*)alloc((size_t)N * 128 * 2);
  _Float16* node2h  = (_Float16*)alloc((size_t)N * 128 * 2);
  _Float16* epre    = (_Float16*)alloc((size_t)E * 128 * 2); // shared by m6 / m8
  float*    aggf    = (float*)alloc((size_t)N * 128 * 4);
  _Float16* aggh    = (_Float16*)alloc((size_t)N * 128 * 2);
  float*    ssum    = (float*)alloc(2 * 128 * 4);
  float*    ssq     = ssum + 128;
  float*    scl     = (float*)alloc(2 * 128 * 4);
  float*    shf     = scl + 128;
  float*    w3f     = (float*)alloc(128 * 4 * 4);
  float*    b3f     = (float*)alloc(4 * 4);
  float*    zkb     = (float*)alloc((size_t)E * 3 * 4);
  float*    deg     = (float*)alloc((size_t)N * 4);
  _Float16* hbuf    = (_Float16*)alloc((size_t)N * 128 * 2);
  float*    inp     = (float*)alloc((size_t)N * 6 * 4);
  _Float16* xob     = (_Float16*)alloc((size_t)N * 48 * 2);
  float*    magg    = (float*)alloc((size_t)N * 128 * 4);
  _Float16* mhb     = (_Float16*)alloc((size_t)N * 128 * 2);

  // ---- weight swizzle (fp32 -> f16, fragment-major for b128 fragment loads) ----
  auto swz = [&](const float* s, _Float16* d, int kreal, int ktot, int ncols) {
    int tot = (ktot >> 5) * (ncols >> 4) * 32;
    k_swizzle_b<<<(tot + 255) / 256, 256, 0, stream>>>(s, d, kreal, ktot, ncols);
  };
  swz(P[0],  wh_enc1, 48, 64, 128);
  swz(P[2],  wh_enc2, 128, 128, 128);
  swz(P[6],  wh_m61,  256, 256, 128);
  swz(P[8],  wh_m62,  128, 128, 128);
  swz(P[12], wh_m71,  128, 128, 128);
  swz(P[14], wh_m72,  128, 128, 128);
  swz(P[18], wh_m81,  512, 512, 128);
  swz(P[20], wh_m82,  128, 128, 128);
  for (int k = 0; k < 3; ++k) {
    swz(P[30] + (size_t)k * 96 * 128,  wh_e1 + (size_t)k * 96 * 128,  96, 96, 128);
    swz(P[32] + (size_t)k * 128 * 128, wh_e2 + (size_t)k * 128 * 128, 128, 128, 128);
  }
  swz(P[28], wh_rec, 128, 128, 48);
  swz(P[36], wh_gwh, 128, 128, 384);
  k_f32_to_f16<<<(N * 48 + 255) / 256, 256, 0, stream>>>(x, xh, N * 48);

  const int TB = 256;
  int nodeBlocks = (N / 16 + 7) / 8;  // 625 tiles
  int edgeBlocks = (E / 16 + 7) / 8;  // 10000 tiles

  // ---- encoder: h5 = BN(MLP2(x)) ----
  hipMemsetAsync(ssum, 0, 2 * 128 * 4, stream);
  k_mlp2<48><<<nodeBlocks, TB, 0, stream>>>(xh, 48, wh_enc1, P[1], wh_enc2, P[3], h5h, ssum,
                                            ssq, N);
  k_bn_finalize<<<1, 128, 0, stream>>>(ssum, ssq, P[4], P[5], (float)N, scl, shf);
  k_bn_apply<<<(N * 128 + 255) / 256, 256, 0, stream>>>(h5h, scl, shf, N * 128);

  // ---- m6: msg = BN(MLP2([h5[tgt], h5[src]])) -> segment_sum ----
  hipMemsetAsync(ssum, 0, 2 * 128 * 4, stream);
  k_edge_mlp<2><<<edgeBlocks, TB, 0, stream>>>(h5h, tgt, h5h, src, nullptr, nullptr, nullptr,
                                               nullptr, wh_m61, P[7], wh_m62, P[9], epre, ssum,
                                               ssq, E);
  k_bn_finalize<<<1, 128, 0, stream>>>(ssum, ssq, P[10], P[11], (float)E, scl, shf);
  hipMemsetAsync(aggf, 0, (size_t)N * 128 * 4, stream);
  k_msg_scatter<<<(int)(((size_t)E * 128 + 255) / 256), 256, 0, stream>>>(epre, scl, shf, tgt,
                                                                          aggf, E);

  // ---- m7: node2 = BN(MLP2(agg)) ----
  k_f32_to_f16<<<(N * 128 + 255) / 256, 256, 0, stream>>>(aggf, aggh, N * 128);
  hipMemsetAsync(ssum, 0, 2 * 128 * 4, stream);
  k_mlp2<128><<<nodeBlocks, TB, 0, stream>>>(aggh, 128, wh_m71, P[13], wh_m72, P[15], node2h,
                                             ssum, ssq, N);
  k_bn_finalize<<<1, 128, 0, stream>>>(ssum, ssq, P[16], P[17], (float)N, scl, shf);
  k_bn_apply<<<(N * 128 + 255) / 256, 256, 0, stream>>>(node2h, scl, shf, N * 128);

  // ---- m8: h8 = BN(MLP2([node2[src], node2[tgt], h5[src], h5[tgt]])) -> zk ----
  hipMemsetAsync(ssum, 0, 2 * 128 * 4, stream);
  k_edge_mlp<4><<<edgeBlocks, TB, 0, stream>>>(node2h, src, node2h, tgt, h5h, src, h5h, tgt,
                                               wh_m81, P[19], wh_m82, P[21], epre, ssum, ssq, E);
  k_bn_finalize<<<1, 128, 0, stream>>>(ssum, ssq, P[22], P[23], (float)E, scl, shf);
  k_fold_w3<<<1, 128, 0, stream>>>(P[24], P[25], scl, shf, w3f, b3f);
  k_logits_zk<<<(E + 255) / 256, 256, 0, stream>>>(epre, w3f, b3f, gum, zkb, E);

  // ---- deg, h init ----
  hipMemsetAsync(deg, 0, (size_t)N * 4, stream);
  k_deg<<<(E + 255) / 256, 256, 0, stream>>>(tgt, deg, E);
  hipMemsetAsync(hbuf, 0, (size_t)N * 128 * 2, stream);

  // ---- recurrence: 8 steps ----
  for (int t = 0; t < 8; ++t) {
    k_prep_inp<<<(N * 6 + 255) / 256, 256, 0, stream>>>(x, out, inp, t, N);
    k_xo<<<nodeBlocks, TB, 0, stream>>>(hbuf, wh_rec, P[29], xob, N);
    hipMemsetAsync(magg, 0, (size_t)N * 128 * 4, stream);
    k_edge_recur<<<edgeBlocks, TB, 0, stream>>>(xob, src, tgt, wh_e1, P[31], wh_e2, P[33], zkb,
                                                magg, E);
    k_mh<<<(N * 128 + 255) / 256, 256, 0, stream>>>(magg, deg, mhb, N * 128);
    k_gru<<<nodeBlocks, TB, 0, stream>>>(mhb, wh_gwh, P[37], P[34], P[35], inp, hbuf, N);
    k_mu<<<(N * 6 + 255) / 256, 256, 0, stream>>>(hbuf, P[26], P[27], inp, out, t, N);
  }
}
